// MinimalEventMamba_5935644803160
// MI455X (gfx1250) — matrix operand order repaired and checked by reading, not compiled, not run on hardware
//
#include <hip/hip_runtime.h>
#include <hip/hip_bf16.h>
#include <math.h>

// ---------------------------------------------------------------------------
// MinimalEventMamba for MI455X (gfx1250, wave32, WMMA)
// B=4, NB=5, HD=64, DI=128, DS=16, DTR=4, DC=4, NL=4, NF=10, H=W=64, L=4096
// All GEMM-shaped work -> v_wmma_f32_16x16x32_bf16 with pre-packed bf16
// operands (no per-element guards in GEMM inner loops).
// ---------------------------------------------------------------------------

#define BATCH 4
#define NB    5
#define HDIM  64
#define DI    128
#define DS    16
#define DTR   4
#define DC    4
#define NL    4
#define NF    10
#define LSEQ  4096
#define MTOT  (BATCH * LSEQ)   // 16384

typedef __attribute__((ext_vector_type(16))) __bf16 v16bf;
typedef __attribute__((ext_vector_type(8)))  __bf16 v8bf;
typedef __attribute__((ext_vector_type(8)))  float  v8f;

__device__ __forceinline__ v8f wmma_bf16(v16bf a, v16bf b, v8f c) {
  return __builtin_amdgcn_wmma_f32_16x16x32_bf16(false, a, false, b, (short)0, c,
                                                 false, false);
}

// ---------------------------------------------------------------------------
// Fragment loaders (CDNA5 ISA 7.12.2 layouts, wave32) — all unconditional,
// contiguous, 16B-aligned vector loads.
// ---------------------------------------------------------------------------

// A fragment from row-major f32 (lda mult of 8, kbase mult of 32).
__device__ __forceinline__ v16bf load_a_f32(const float* __restrict__ A, int lda,
                                            int tileM, int kbase) {
  int lane = threadIdx.x & 31;
  const float* ap = A + (tileM + (lane & 15)) * lda + kbase + ((lane >> 4) << 3);
  v16bf a;
#pragma unroll
  for (int e = 0; e < 16; ++e) {
    int kk = ((e & 8) << 1) + (e & 7);   // 0..7 then 16..23
    a[e] = (__bf16)ap[kk];
  }
  return a;
}

// A fragment from row-major bf16 (lda mult of 8, kbase mult of 32).
__device__ __forceinline__ v16bf load_a_bf16(const __bf16* __restrict__ A, int lda,
                                             int tileM, int kbase) {
  int lane = threadIdx.x & 31;
  const __bf16* ap = A + (tileM + (lane & 15)) * lda + kbase + ((lane >> 4) << 3);
  v8bf lo = *(const v8bf*)ap;          // K offsets 0..7
  v8bf hi = *(const v8bf*)(ap + 16);   // K offsets 16..23
  v16bf a;
#pragma unroll
  for (int e = 0; e < 8; ++e) { a[e] = lo[e]; a[8 + e] = hi[e]; }
  return a;
}

// B fragment (= W^T tile) from pre-packed bf16 weights [Npad][Kpad].
__device__ __forceinline__ v16bf load_b_packed(const __bf16* __restrict__ Bp,
                                               int Kpad, int tileN, int kbase) {
  int lane = threadIdx.x & 31;
  int n  = tileN + (lane & 15);
  int kb = kbase + ((lane >> 4) << 3);
  const v8bf* p = (const v8bf*)(Bp + n * Kpad + kb);
  v8bf lo = p[0];
  v8bf hi = p[2];
  v16bf b;
#pragma unroll
  for (int e = 0; e < 8; ++e) { b[e] = lo[e]; b[8 + e] = hi[e]; }
  return b;
}

__device__ __forceinline__ void store_tile(float* __restrict__ C, int ldc,
                                           int tileM, int tileN, v8f acc) {
  int lane = threadIdx.x & 31;
  int n  = tileN + (lane & 15);
  int m0 = tileM + ((lane >> 4) << 3);
#pragma unroll
  for (int v = 0; v < 8; ++v) C[(m0 + v) * ldc + n] = acc[v];
}

// ---------------------------------------------------------------------------
// Operand packing kernels
// ---------------------------------------------------------------------------

// W (N x K f32) -> zero-padded bf16 [Npad][Kpad]
__global__ void pack_w(const float* __restrict__ W, int N, int K,
                       __bf16* __restrict__ out, int Kpad, int total) {
  int idx = blockIdx.x * 256 + threadIdx.x;
  if (idx >= total) return;
  int n = idx / Kpad, k = idx - n * Kpad;
  float v = (n < N && k < K) ? W[n * K + k] : 0.0f;
  out[idx] = (__bf16)v;
}

// im2col for 3x3 SAME conv -> bf16 [MTOT][Kpad], k = ci*9 + tap, zero pad.
// layout 0: input NCHW; layout 1: input NLC.
__global__ void im2col_k(const float* __restrict__ in, __bf16* __restrict__ A,
                         int CIN, int KTOT, int Kpad, int layout) {
  int idx = blockIdx.x * 256 + threadIdx.x;   // over MTOT * Kpad
  int m = idx / Kpad, k = idx - m * Kpad;
  int b = m >> 12, p = m & 4095, hh = p >> 6, ww = p & 63;
  float v = 0.0f;
  if (k < KTOT) {
    int ci = k / 9, tap = k - ci * 9;
    int dh = tap / 3 - 1, dw = tap - (tap / 3) * 3 - 1;
    int h2 = hh + dh, w2 = ww + dw;
    if ((unsigned)h2 < 64u && (unsigned)w2 < 64u)
      v = (layout == 0) ? in[((b * CIN + ci) << 12) + (h2 << 6) + w2]
                        : in[(((b << 12) + (h2 << 6) + w2)) * CIN + ci];
  }
  A[idx] = (__bf16)v;
}

// ---------------------------------------------------------------------------
// GEMM kernels. One wave computes a 16 x (16*NR) strip. 8 waves / block.
// ---------------------------------------------------------------------------

// f32 activations x packed-W^T -> row-major f32 out (mamba in_proj / x_proj)
template <int NR>
__global__ void gemm_rm_af32(const float* __restrict__ A, int lda,
                             const __bf16* __restrict__ Bp, int Kpad, int K,
                             float* __restrict__ out, int ldo, int NTG) {
  int wave   = (blockIdx.x * blockDim.x + threadIdx.x) >> 5;
  int tileM  = (wave / NTG) << 4;
  int tileN0 = (wave % NTG) * (16 * NR);
  v8f acc[NR];
#pragma unroll
  for (int r = 0; r < NR; ++r) acc[r] = (v8f){0, 0, 0, 0, 0, 0, 0, 0};
  for (int k0 = 0; k0 < K; k0 += 32) {
    v16bf a = load_a_f32(A, lda, tileM, k0);
#pragma unroll
    for (int r = 0; r < NR; ++r) {
      v16bf b = load_b_packed(Bp, Kpad, tileN0 + 16 * r, k0);
      acc[r] = wmma_bf16(a, b, acc[r]);
    }
  }
#pragma unroll
  for (int r = 0; r < NR; ++r)
    store_tile(out, ldo, tileM, tileN0 + 16 * r, acc[r]);
}

// bf16 im2col x packed-W^T -> NCHW f32 out (+bias, guard n < COUT)
template <int NR>
__global__ void gemm_conv(const __bf16* __restrict__ A, int Kpad,
                          const __bf16* __restrict__ Bp,
                          const float* __restrict__ bias, int COUT,
                          float* __restrict__ out) {
  int wave  = (blockIdx.x * blockDim.x + threadIdx.x) >> 5;
  int tileM = wave << 4;                 // NTG == 1
  v8f acc[NR];
#pragma unroll
  for (int r = 0; r < NR; ++r) acc[r] = (v8f){0, 0, 0, 0, 0, 0, 0, 0};
  for (int k0 = 0; k0 < Kpad; k0 += 32) {
    v16bf a = load_a_bf16(A, Kpad, tileM, k0);
#pragma unroll
    for (int r = 0; r < NR; ++r) {
      v16bf b = load_b_packed(Bp, Kpad, 16 * r, k0);
      acc[r] = wmma_bf16(a, b, acc[r]);
    }
  }
  int lane = threadIdx.x & 31;
  int m0   = tileM + ((lane >> 4) << 3);
#pragma unroll
  for (int r = 0; r < NR; ++r) {
    int n = 16 * r + (lane & 15);
    if (n < COUT) {
      float bv = bias[n];
#pragma unroll
      for (int v = 0; v < 8; ++v) {
        int mm = m0 + v;
        int bb = mm >> 12, pp = mm & 4095;
        out[((bb * COUT + n) << 12) + pp] = acc[r][v] + bv;
      }
    }
  }
}

// fused gate + out_proj + residual:  t += (y * silu(z)) @ out_proj^T
template <int NR>
__global__ void gemm_gate_outproj(const float* __restrict__ y,
                                  const float* __restrict__ xz,
                                  const __bf16* __restrict__ Bp,  // [64][128]
                                  float* __restrict__ t) {
  int wave  = (blockIdx.x * blockDim.x + threadIdx.x) >> 5;
  int tileM = wave << 4;                 // NTG == 1
  int lane  = threadIdx.x & 31;
  int mrow  = tileM + (lane & 15);
  int khalf = (lane >> 4) << 3;
  v8f acc[NR];
#pragma unroll
  for (int r = 0; r < NR; ++r) acc[r] = (v8f){0, 0, 0, 0, 0, 0, 0, 0};
  for (int k0 = 0; k0 < DI; k0 += 32) {
    v16bf a;
#pragma unroll
    for (int e = 0; e < 16; ++e) {
      int k = k0 + khalf + ((e & 8) << 1) + (e & 7);
      float yv = y[(mrow << 7) + k];
      float zv = xz[(mrow << 8) + DI + k];
      float g  = zv / (1.0f + __expf(-zv));
      a[e] = (__bf16)(yv * g);
    }
#pragma unroll
    for (int r = 0; r < NR; ++r) {
      v16bf b = load_b_packed(Bp, DI, 16 * r, k0);
      acc[r] = wmma_bf16(a, b, acc[r]);
    }
  }
  int m0 = tileM + ((lane >> 4) << 3);
#pragma unroll
  for (int r = 0; r < NR; ++r) {
    int n = 16 * r + (lane & 15);
#pragma unroll
    for (int v = 0; v < 8; ++v)
      t[(m0 + v) * HDIM + n] += acc[r][v];  // residual
  }
}

// ---------------------------------------------------------------------------
// BatchNorm (training-mode biased stats)
// ---------------------------------------------------------------------------
__global__ void bn_stats(const float* __restrict__ buf, float* __restrict__ stats) {
  int c = blockIdx.x;
  __shared__ float ssum[256];
  __shared__ float ssq[256];
  float s = 0.f, q = 0.f;
  for (int b = 0; b < BATCH; ++b) {
    const float* p = buf + ((b * HDIM + c) << 12);
    for (int i = threadIdx.x; i < 4096; i += 256) {
      float v = p[i];
      s += v; q += v * v;
    }
  }
  ssum[threadIdx.x] = s; ssq[threadIdx.x] = q;
  __syncthreads();
  for (int off = 128; off > 0; off >>= 1) {
    if ((int)threadIdx.x < off) {
      ssum[threadIdx.x] += ssum[threadIdx.x + off];
      ssq[threadIdx.x]  += ssq[threadIdx.x + off];
    }
    __syncthreads();
  }
  if (threadIdx.x == 0) {
    float mean = ssum[0] * (1.0f / 16384.0f);
    float var  = ssq[0] * (1.0f / 16384.0f) - mean * mean;
    stats[c]        = mean;
    stats[HDIM + c] = rsqrtf(var + 1e-5f);
  }
}

__global__ void bn_apply(const float* __restrict__ buf,
                         const float* __restrict__ stats,
                         const float* __restrict__ gamma,
                         const float* __restrict__ beta,
                         float* __restrict__ out) {
  int idx = blockIdx.x * 256 + threadIdx.x;      // NCHW order, C=64
  int p = idx & 4095;
  int c = (idx >> 12) & 63;
  int b = idx >> 18;
  float v = buf[idx];
  float y = gamma[c] * (v - stats[c]) * stats[HDIM + c] + beta[c];
  out[((b << 12) + p) * HDIM + c] = fmaxf(y, 0.0f);
}

// ---------------------------------------------------------------------------
// Mamba element-wise kernels
// ---------------------------------------------------------------------------
__global__ void dwconv_silu(const float* __restrict__ xz,
                            const float* __restrict__ cw,
                            const float* __restrict__ cb,
                            float* __restrict__ xi) {
  int idx = blockIdx.x * 256 + threadIdx.x;   // 16384*128
  int d   = idx & (DI - 1);
  int row = idx >> 7;
  int l   = row & 4095;
  int b   = row >> 12;
  float acc = cb[d];
#pragma unroll
  for (int k = 0; k < DC; ++k) {
    int ls = l + k - (DC - 1);
    if (ls >= 0) acc += xz[(((b << 12) + ls) << 8) + d] * cw[d * DC + k];
  }
  xi[idx] = acc / (1.0f + __expf(-acc));
}

__global__ void dt_softplus(const float* __restrict__ xdbc,
                            const float* __restrict__ dtw,
                            const float* __restrict__ dtb,
                            float* __restrict__ delta) {
  int idx = blockIdx.x * 256 + threadIdx.x;   // 16384*128
  int d   = idx & (DI - 1);
  int row = idx >> 7;
  const float* dt = xdbc + row * 48;
  float acc = dtb[d];
#pragma unroll
  for (int r = 0; r < DTR; ++r) acc += dt[r] * dtw[d * DTR + r];
  delta[idx] = (acc > 20.0f) ? acc : log1pf(__expf(acc));
}

// ---------------------------------------------------------------------------
// Selective scan: block per batch, thread per channel, 16 states in registers.
// ---------------------------------------------------------------------------
__global__ void selective_scan(const float* __restrict__ delta,
                               const float* __restrict__ u,
                               const float* __restrict__ xdbc,
                               const float* __restrict__ A_log,
                               const float* __restrict__ Dp,
                               float* __restrict__ y) {
  int b = blockIdx.x;
  int d = threadIdx.x;
  float a[DS];
#pragma unroll
  for (int s = 0; s < DS; ++s) a[s] = -__expf(A_log[d * DS + s]);
  float Dv = Dp[d];
  float h[DS];
#pragma unroll
  for (int s = 0; s < DS; ++s) h[s] = 0.0f;

  int base = b << 12;
  for (int t = 0; t < LSEQ; ++t) {
    int row = base + t;
    float dt = delta[(row << 7) + d];
    float ut = u[(row << 7) + d];
    float du = dt * ut;
    const float* bc = xdbc + row * 48 + DTR;
    float yv = 0.0f;
#pragma unroll
    for (int s = 0; s < DS; ++s) {
      float dA = __expf(dt * a[s]);
      h[s] = dA * h[s] + du * bc[s];
      yv  += h[s] * bc[DS + s];
    }
    y[(row << 7) + d] = yv + ut * Dv;
  }
}

// ---------------------------------------------------------------------------
// Host launch
// ---------------------------------------------------------------------------
extern "C" void kernel_launch(void* const* d_in, const int* in_sizes, int n_in,
                              void* d_out, int out_size, void* d_ws, size_t ws_size,
                              hipStream_t stream) {
  const float* x       = (const float*)d_in[0];
  const float* enc_w   = (const float*)d_in[1];
  const float* enc_b   = (const float*)d_in[2];
  const float* enc_g   = (const float*)d_in[3];
  const float* enc_be  = (const float*)d_in[4];
  const float* in_proj = (const float*)d_in[5];
  const float* conv_w  = (const float*)d_in[6];
  const float* conv_b  = (const float*)d_in[7];
  const float* x_proj  = (const float*)d_in[8];
  const float* dt_w    = (const float*)d_in[9];
  const float* dt_b    = (const float*)d_in[10];
  const float* A_log   = (const float*)d_in[11];
  const float* Dp      = (const float*)d_in[12];
  const float* out_proj= (const float*)d_in[13];
  const float* dec1_w  = (const float*)d_in[14];
  const float* dec1_b  = (const float*)d_in[15];
  const float* dec1_g  = (const float*)d_in[16];
  const float* dec1_be = (const float*)d_in[17];
  const float* dec2_w  = (const float*)d_in[18];
  const float* dec2_b  = (const float*)d_in[19];
  float* out = (float*)d_out;
  (void)in_sizes; (void)n_in; (void)out_size; (void)ws_size;

  // ---- workspace layout ----
  float* ws    = (float*)d_ws;
  float* bufA  = ws;                 // 1,048,576  conv out NCHW
  float* tbuf  = bufA  + 1048576;    // 1,048,576  t (B,L,64)
  float* xz    = tbuf  + 1048576;    // 4,194,304  (B,L,256)
  float* xi    = xz    + 4194304;    // 2,097,152  (B,L,128)
  float* xdbc  = xi    + 2097152;    //   786,432  (B,L,48) padded
  float* delta = xdbc  + 786432;     // 2,097,152  (B,L,128)
  float* ybuf  = delta + 2097152;    // 2,097,152  (B,L,128)
  float* stats = ybuf  + 2097152;    //       128
  __bf16* bb   = (__bf16*)(stats + 128);
  __bf16* Acol = bb;                 // 9,437,184  im2col (max 16384x576)
  __bf16* Wenc = Acol + 9437184;     //     4,096  64x64
  __bf16* Wd1  = Wenc + 4096;        //    36,864  64x576
  __bf16* Wd2  = Wd1  + 36864;       //     9,216  16x576
  __bf16* Win  = Wd2  + 9216;        //    65,536  4 x 256x64
  __bf16* Wxp  = Win  + 65536;       //    24,576  4 x 48x128
  __bf16* Wop  = Wxp  + 24576;       //    32,768  4 x 64x128

  dim3 blk(256);

  // ---- pack all weights to padded bf16 ----
  pack_w<<<16, blk, 0, stream>>>(enc_w, HDIM, NB * 9, Wenc, 64, 64 * 64);
  pack_w<<<144, blk, 0, stream>>>(dec1_w, HDIM, HDIM * 9, Wd1, 576, 64 * 576);
  pack_w<<<36, blk, 0, stream>>>(dec2_w, NF, HDIM * 9, Wd2, 576, 16 * 576);
  for (int i = 0; i < NL; ++i) {
    pack_w<<<64, blk, 0, stream>>>(in_proj + (size_t)i * 2 * DI * HDIM,
                                   2 * DI, HDIM, Win + i * 16384, HDIM, 16384);
    pack_w<<<24, blk, 0, stream>>>(x_proj + (size_t)i * 36 * DI,
                                   36, DI, Wxp + i * 6144, DI, 6144);
    pack_w<<<32, blk, 0, stream>>>(out_proj + (size_t)i * HDIM * DI,
                                   HDIM, DI, Wop + i * 8192, DI, 8192);
  }

  // ---- encoder: im2col + WMMA GEMM (K=45 pad 64, N=64) ----
  im2col_k<<<4096, blk, 0, stream>>>(x, Acol, NB, NB * 9, 64, /*layout=*/0);
  gemm_conv<4><<<128, blk, 0, stream>>>(Acol, 64, Wenc, enc_b, HDIM, bufA);
  bn_stats<<<HDIM, blk, 0, stream>>>(bufA, stats);
  bn_apply<<<4096, blk, 0, stream>>>(bufA, stats, enc_g, enc_be, tbuf);

  // ---- 4 Mamba layers ----
  for (int i = 0; i < NL; ++i) {
    const float* cwW = conv_w + (size_t)i * DI * DC;
    const float* cbW = conv_b + (size_t)i * DI;
    const float* dwW = dt_w   + (size_t)i * DI * DTR;
    const float* dbW = dt_b   + (size_t)i * DI;
    const float* alW = A_log  + (size_t)i * DI * DS;
    const float* dpW = Dp     + (size_t)i * DI;

    // xz = t @ in_proj^T : M=16384, N=256, K=64  (NR=4, NTG=4)
    gemm_rm_af32<4><<<512, blk, 0, stream>>>(tbuf, HDIM, Win + i * 16384, HDIM,
                                             HDIM, xz, 2 * DI, 4);
    dwconv_silu<<<8192, blk, 0, stream>>>(xz, cwW, cbW, xi);
    // xdbc = xi @ x_proj^T : M=16384, N=48(pad), K=128 (NR=3, NTG=1)
    gemm_rm_af32<3><<<128, blk, 0, stream>>>(xi, DI, Wxp + i * 6144, DI,
                                             DI, xdbc, 48, 1);
    dt_softplus<<<8192, blk, 0, stream>>>(xdbc, dwW, dbW, delta);
    selective_scan<<<BATCH, dim3(DI), 0, stream>>>(delta, xi, xdbc, alW, dpW, ybuf);
    // t += (y * silu(z)) @ out_proj^T : N=64 (NR=4)
    gemm_gate_outproj<4><<<128, blk, 0, stream>>>(ybuf, xz, Wop + i * 8192, tbuf);
  }

  // ---- decoder ----
  im2col_k<<<36864, blk, 0, stream>>>(tbuf, Acol, HDIM, HDIM * 9, 576, /*layout=*/1);
  gemm_conv<4><<<128, blk, 0, stream>>>(Acol, 576, Wd1, dec1_b, HDIM, bufA);
  bn_stats<<<HDIM, blk, 0, stream>>>(bufA, stats);
  bn_apply<<<4096, blk, 0, stream>>>(bufA, stats, dec1_g, dec1_be, tbuf);
  im2col_k<<<36864, blk, 0, stream>>>(tbuf, Acol, HDIM, HDIM * 9, 576, /*layout=*/1);
  gemm_conv<1><<<128, blk, 0, stream>>>(Acol, 576, Wd2, dec2_b, NF, out);
}
